// GnBlock_35553739276321
// MI455X (gfx1250) — matrix operand order repaired and checked by reading, not compiled, not run on hardware
//
#include <hip/hip_runtime.h>
#include <hip/hip_bf16.h>

#define NNODE 50000
#define NEDGE 800000
#define DLAT 128
#define LN_EPS 1e-5f

typedef __bf16 bf16_t;
typedef __attribute__((ext_vector_type(16))) __bf16 v16bf;
typedef __attribute__((ext_vector_type(8)))  __bf16 v8bf;
typedef __attribute__((ext_vector_type(8)))  float  v8f;
typedef __attribute__((ext_vector_type(4)))  float  v4f;

// ---------------------------------------------------------------------------
// WMMA helper: D = A(16x32 bf16) * B(32x16 bf16) + C(16x16 f32)
// ---------------------------------------------------------------------------
__device__ __forceinline__ v8f wmma_bf16(v16bf a, v16bf b, v8f c) {
  return __builtin_amdgcn_wmma_f32_16x16x32_bf16(
      /*neg_a=*/false, a, /*neg_b=*/false, b,
      /*c_mod=*/(short)0, c, /*reuse_a=*/false, /*reuse_b=*/false);
}

// A-matrix tile from an fp32 row (row-major source, convert to bf16).
// ISA 16-bit A 16x32 layout: lanes 0-15 -> K = klo..klo+7 (VGPR0-3) and
// klo+16..klo+23 (VGPR4-7); lanes 16-31 shifted by +8 (klo already adjusted).
__device__ __forceinline__ v16bf load_a_f32(const float* __restrict__ rowp,
                                            int klo) {
  const v4f* p0 = (const v4f*)(rowp + klo);
  const v4f* p1 = (const v4f*)(rowp + klo + 16);
  v4f f0 = p0[0], f1 = p0[1];
  v4f f2 = p1[0], f3 = p1[1];
  v16bf r;
#pragma unroll
  for (int i = 0; i < 4; ++i) {
    r[i]      = (bf16_t)f0[i];
    r[i + 4]  = (bf16_t)f1[i];
    r[i + 8]  = (bf16_t)f2[i];
    r[i + 12] = (bf16_t)f3[i];
  }
  return r;
}

// A-matrix tile from a bf16 row already in LDS.
__device__ __forceinline__ v16bf load_a_bf16(const bf16_t* rowp, int klo) {
  v8bf lo = *(const v8bf*)(rowp + klo);
  v8bf hi = *(const v8bf*)(rowp + klo + 16);
  return __builtin_shufflevector(lo, hi, 0, 1, 2, 3, 4, 5, 6, 7,
                                 8, 9, 10, 11, 12, 13, 14, 15);
}

// B-matrix tile from transposed weights Wt[n][k] (bf16, row length ldk).
// Lane n holds K = k..k+7 and k+16..k+23 (k already includes +8 for lanes>=16).
__device__ __forceinline__ v16bf load_b16(const bf16_t* __restrict__ Wt,
                                          int ldk, int n, int k) {
  const bf16_t* p = Wt + (size_t)n * ldk + k;
  v8bf lo = *(const v8bf*)p;
  v8bf hi = *(const v8bf*)(p + 16);
  return __builtin_shufflevector(lo, hi, 0, 1, 2, 3, 4, 5, 6, 7,
                                 8, 9, 10, 11, 12, 13, 14, 15);
}

__device__ __forceinline__ void zero_acc(v8f* acc) {
#pragma unroll
  for (int nt = 0; nt < 8; ++nt) {
    v8f z = {0.f, 0.f, 0.f, 0.f, 0.f, 0.f, 0.f, 0.f};
    acc[nt] = z;
  }
}

// ---------------------------------------------------------------------------
// Weight transpose + bf16 convert: dst[n*K + k] = (bf16) src[k*128 + n]
// ---------------------------------------------------------------------------
__global__ void transpose_bf16_kernel(const float* __restrict__ src,
                                      bf16_t* __restrict__ dst, int K) {
  int t = blockIdx.x * blockDim.x + threadIdx.x;
  if (t >= K * DLAT) return;
  int n = t / K;
  int k = t - n * K;
  dst[t] = (bf16_t)src[(size_t)k * DLAT + n];
}

__global__ void zero_kernel(float* __restrict__ p, int n) {
  int t = blockIdx.x * blockDim.x + threadIdx.x;
  if (t < n) p[t] = 0.f;
}

// ---------------------------------------------------------------------------
// Edge block: h = LN(MLP([x[s], x[r], e_attr])); out_edge = h; agg[r] += h
// 128 threads = 4 waves; wave w owns edges [blk*64 + w*16, +16).
// ---------------------------------------------------------------------------
__global__ __launch_bounds__(128) void edge_mlp_kernel(
    const float* __restrict__ x, const float* __restrict__ edge_attr,
    const int* __restrict__ eidx,
    const bf16_t* __restrict__ W1t, const float* __restrict__ b1,
    const bf16_t* __restrict__ W2t, const float* __restrict__ b2,
    const bf16_t* __restrict__ W3t, const float* __restrict__ b3,
    const float* __restrict__ lng, const float* __restrict__ lnb,
    float* __restrict__ out_edge, float* __restrict__ agg) {
  __shared__ bf16_t hA[4 * 16 * DLAT];  // 16 KB : h1 (bf16)
  __shared__ bf16_t hB[4 * 16 * DLAT];  // 16 KB : h2 (bf16)
  __shared__ float  hF[4 * 16 * DLAT];  // 32 KB : h3 (f32, pre-LN)

  const int tid  = threadIdx.x;
  const int w    = tid >> 5;
  const int lane = tid & 31;
  const int row  = lane & 15;
  const int half = lane >> 4;
  const int nlo  = lane & 15;
  const int kA   = half * 8;    // A-layout lane K shift
  const int kB   = half * 8;    // B-layout lane K shift

  const int eBlk = blockIdx.x * 64;
  const int e    = eBlk + w * 16 + row;
  const int s    = eidx[e];
  const int r    = eidx[NEDGE + e];
  const float* xs = x + (size_t)s * DLAT;
  const float* xr = x + (size_t)r * DLAT;
  const float* ea = edge_attr + (size_t)e * DLAT;

  bf16_t* hArow = hA + w * 16 * DLAT;
  bf16_t* hBrow = hB + w * 16 * DLAT;
  float*  hFrow = hF + w * 16 * DLAT;

  v8f acc[8];

  // ---- layer 1: [16,384] @ [384,128] ----
  zero_acc(acc);
#pragma unroll
  for (int c = 0; c < 12; ++c) {
    const float* rowp = (c < 4) ? xs : (c < 8) ? xr : ea;
    int k0 = (c & 3) * 32;
    v16bf a = load_a_f32(rowp, k0 + kA);
#pragma unroll
    for (int nt = 0; nt < 8; ++nt) {
      v16bf b = load_b16(W1t, 384, nt * 16 + nlo, c * 32 + kB);
      acc[nt] = wmma_bf16(a, b, acc[nt]);
    }
  }
#pragma unroll
  for (int nt = 0; nt < 8; ++nt) {
    int n = nt * 16 + nlo;
    float bn = b1[n];
#pragma unroll
    for (int j = 0; j < 8; ++j) {
      float v = acc[nt][j] + bn;
      hArow[(j + 8 * half) * DLAT + n] = (bf16_t)(v > 0.f ? v : 0.f);
    }
  }

  // ---- layer 2: [16,128] @ [128,128] ----
  zero_acc(acc);
#pragma unroll
  for (int c = 0; c < 4; ++c) {
    v16bf a = load_a_bf16(hArow + row * DLAT, c * 32 + kA);
#pragma unroll
    for (int nt = 0; nt < 8; ++nt) {
      v16bf b = load_b16(W2t, 128, nt * 16 + nlo, c * 32 + kB);
      acc[nt] = wmma_bf16(a, b, acc[nt]);
    }
  }
#pragma unroll
  for (int nt = 0; nt < 8; ++nt) {
    int n = nt * 16 + nlo;
    float bn = b2[n];
#pragma unroll
    for (int j = 0; j < 8; ++j) {
      float v = acc[nt][j] + bn;
      hBrow[(j + 8 * half) * DLAT + n] = (bf16_t)(v > 0.f ? v : 0.f);
    }
  }

  // ---- layer 3: [16,128] @ [128,128] (no relu) ----
  zero_acc(acc);
#pragma unroll
  for (int c = 0; c < 4; ++c) {
    v16bf a = load_a_bf16(hBrow + row * DLAT, c * 32 + kA);
#pragma unroll
    for (int nt = 0; nt < 8; ++nt) {
      v16bf b = load_b16(W3t, 128, nt * 16 + nlo, c * 32 + kB);
      acc[nt] = wmma_bf16(a, b, acc[nt]);
    }
  }
#pragma unroll
  for (int nt = 0; nt < 8; ++nt) {
    int n = nt * 16 + nlo;
    float bn = b3[n];
#pragma unroll
    for (int j = 0; j < 8; ++j)
      hFrow[(j + 8 * half) * DLAT + n] = acc[nt][j] + bn;
  }

  __syncthreads();

  // ---- LayerNorm + write + scatter-add (one thread per edge row) ----
  if (tid < 64) {
    const float* hr = hF + tid * DLAT;
    float mu = 0.f;
#pragma unroll 4
    for (int c = 0; c < DLAT; ++c) mu += hr[c];
    mu *= (1.f / DLAT);
    float var = 0.f;
#pragma unroll 4
    for (int c = 0; c < DLAT; ++c) {
      float d = hr[c] - mu;
      var += d * d;
    }
    var *= (1.f / DLAT);
    float inv = __frsqrt_rn(var + LN_EPS);

    int eg = eBlk + tid;
    int rcv = eidx[NEDGE + eg];
    float* aggrow = agg + (size_t)rcv * DLAT;
    float* orow   = out_edge + (size_t)eg * DLAT;
#pragma unroll 4
    for (int c = 0; c < DLAT; ++c) {
      float v = (hr[c] - mu) * inv * lng[c] + lnb[c];
      orow[c] = v;
      atomicAdd(aggrow + c, v);
    }
  }
}

// ---------------------------------------------------------------------------
// Node block: x_out = x + LN(MLP([x, agg]))
// ---------------------------------------------------------------------------
__global__ __launch_bounds__(128) void node_mlp_kernel(
    const float* __restrict__ x, const float* __restrict__ agg,
    const bf16_t* __restrict__ W1t, const float* __restrict__ b1,
    const bf16_t* __restrict__ W2t, const float* __restrict__ b2,
    const bf16_t* __restrict__ W3t, const float* __restrict__ b3,
    const float* __restrict__ lng, const float* __restrict__ lnb,
    float* __restrict__ x_out) {
  __shared__ bf16_t hA[4 * 16 * DLAT];
  __shared__ bf16_t hB[4 * 16 * DLAT];
  __shared__ float  hF[4 * 16 * DLAT];

  const int tid  = threadIdx.x;
  const int w    = tid >> 5;
  const int lane = tid & 31;
  const int row  = lane & 15;
  const int half = lane >> 4;
  const int nlo  = lane & 15;
  const int kA   = half * 8;
  const int kB   = half * 8;

  const int nBlk = blockIdx.x * 64;
  int nd = nBlk + w * 16 + row;
  int nc = nd < NNODE ? nd : (NNODE - 1);  // clamp: keep EXEC all-ones
  const float* xn = x + (size_t)nc * DLAT;
  const float* an = agg + (size_t)nc * DLAT;

  bf16_t* hArow = hA + w * 16 * DLAT;
  bf16_t* hBrow = hB + w * 16 * DLAT;
  float*  hFrow = hF + w * 16 * DLAT;

  v8f acc[8];

  // ---- layer 1: [16,256] @ [256,128] ----
  zero_acc(acc);
#pragma unroll
  for (int c = 0; c < 8; ++c) {
    const float* rowp = (c < 4) ? xn : an;
    int k0 = (c & 3) * 32;
    v16bf a = load_a_f32(rowp, k0 + kA);
#pragma unroll
    for (int nt = 0; nt < 8; ++nt) {
      v16bf b = load_b16(W1t, 256, nt * 16 + nlo, c * 32 + kB);
      acc[nt] = wmma_bf16(a, b, acc[nt]);
    }
  }
#pragma unroll
  for (int nt = 0; nt < 8; ++nt) {
    int n = nt * 16 + nlo;
    float bn = b1[n];
#pragma unroll
    for (int j = 0; j < 8; ++j) {
      float v = acc[nt][j] + bn;
      hArow[(j + 8 * half) * DLAT + n] = (bf16_t)(v > 0.f ? v : 0.f);
    }
  }

  // ---- layer 2 ----
  zero_acc(acc);
#pragma unroll
  for (int c = 0; c < 4; ++c) {
    v16bf a = load_a_bf16(hArow + row * DLAT, c * 32 + kA);
#pragma unroll
    for (int nt = 0; nt < 8; ++nt) {
      v16bf b = load_b16(W2t, 128, nt * 16 + nlo, c * 32 + kB);
      acc[nt] = wmma_bf16(a, b, acc[nt]);
    }
  }
#pragma unroll
  for (int nt = 0; nt < 8; ++nt) {
    int n = nt * 16 + nlo;
    float bn = b2[n];
#pragma unroll
    for (int j = 0; j < 8; ++j) {
      float v = acc[nt][j] + bn;
      hBrow[(j + 8 * half) * DLAT + n] = (bf16_t)(v > 0.f ? v : 0.f);
    }
  }

  // ---- layer 3 (no relu) ----
  zero_acc(acc);
#pragma unroll
  for (int c = 0; c < 4; ++c) {
    v16bf a = load_a_bf16(hBrow + row * DLAT, c * 32 + kA);
#pragma unroll
    for (int nt = 0; nt < 8; ++nt) {
      v16bf b = load_b16(W3t, 128, nt * 16 + nlo, c * 32 + kB);
      acc[nt] = wmma_bf16(a, b, acc[nt]);
    }
  }
#pragma unroll
  for (int nt = 0; nt < 8; ++nt) {
    int n = nt * 16 + nlo;
    float bn = b3[n];
#pragma unroll
    for (int j = 0; j < 8; ++j)
      hFrow[(j + 8 * half) * DLAT + n] = acc[nt][j] + bn;
  }

  __syncthreads();

  // ---- LayerNorm + residual ----
  if (tid < 64) {
    int ng = nBlk + tid;
    if (ng < NNODE) {
      const float* hr = hF + tid * DLAT;
      float mu = 0.f;
#pragma unroll 4
      for (int c = 0; c < DLAT; ++c) mu += hr[c];
      mu *= (1.f / DLAT);
      float var = 0.f;
#pragma unroll 4
      for (int c = 0; c < DLAT; ++c) {
        float d = hr[c] - mu;
        var += d * d;
      }
      var *= (1.f / DLAT);
      float inv = __frsqrt_rn(var + LN_EPS);
      const float* xrow = x + (size_t)ng * DLAT;
      float* orow = x_out + (size_t)ng * DLAT;
#pragma unroll 4
      for (int c = 0; c < DLAT; ++c)
        orow[c] = xrow[c] + (hr[c] - mu) * inv * lng[c] + lnb[c];
    }
  }
}

// ---------------------------------------------------------------------------
extern "C" void kernel_launch(void* const* d_in, const int* in_sizes, int n_in,
                              void* d_out, int out_size, void* d_ws,
                              size_t ws_size, hipStream_t stream) {
  (void)in_sizes; (void)n_in; (void)out_size; (void)ws_size;

  const float* x         = (const float*)d_in[0];
  const float* edge_attr = (const float*)d_in[1];
  const int*   eidx      = (const int*)d_in[2];
  const float* eb_w1 = (const float*)d_in[3];
  const float* eb_b1 = (const float*)d_in[4];
  const float* eb_w2 = (const float*)d_in[5];
  const float* eb_b2 = (const float*)d_in[6];
  const float* eb_w3 = (const float*)d_in[7];
  const float* eb_b3 = (const float*)d_in[8];
  const float* eb_g  = (const float*)d_in[9];
  const float* eb_be = (const float*)d_in[10];
  const float* nb_w1 = (const float*)d_in[11];
  const float* nb_b1 = (const float*)d_in[12];
  const float* nb_w2 = (const float*)d_in[13];
  const float* nb_b2 = (const float*)d_in[14];
  const float* nb_w3 = (const float*)d_in[15];
  const float* nb_b3 = (const float*)d_in[16];
  const float* nb_g  = (const float*)d_in[17];
  const float* nb_be = (const float*)d_in[18];

  unsigned char* ws = (unsigned char*)d_ws;
  bf16_t* ebW1t = (bf16_t*)(ws + 0);        // 128*384*2 = 98304
  bf16_t* ebW2t = (bf16_t*)(ws + 98304);    // +32768
  bf16_t* ebW3t = (bf16_t*)(ws + 131072);   // +32768
  bf16_t* nbW1t = (bf16_t*)(ws + 163840);   // 128*256*2 = 65536
  bf16_t* nbW2t = (bf16_t*)(ws + 229376);   // +32768
  bf16_t* nbW3t = (bf16_t*)(ws + 262144);   // +32768
  float*  agg   = (float*)(ws + 294912);    // NNODE*128*4 = 25.6 MB

  float* x_out    = (float*)d_out;
  float* out_edge = (float*)d_out + (size_t)NNODE * DLAT;

  const int TPB = 256;
  transpose_bf16_kernel<<<(384 * DLAT + TPB - 1) / TPB, TPB, 0, stream>>>(
      eb_w1, ebW1t, 384);
  transpose_bf16_kernel<<<(128 * DLAT + TPB - 1) / TPB, TPB, 0, stream>>>(
      eb_w2, ebW2t, 128);
  transpose_bf16_kernel<<<(128 * DLAT + TPB - 1) / TPB, TPB, 0, stream>>>(
      eb_w3, ebW3t, 128);
  transpose_bf16_kernel<<<(256 * DLAT + TPB - 1) / TPB, TPB, 0, stream>>>(
      nb_w1, nbW1t, 256);
  transpose_bf16_kernel<<<(128 * DLAT + TPB - 1) / TPB, TPB, 0, stream>>>(
      nb_w2, nbW2t, 128);
  transpose_bf16_kernel<<<(128 * DLAT + TPB - 1) / TPB, TPB, 0, stream>>>(
      nb_w3, nbW3t, 128);

  zero_kernel<<<(NNODE * DLAT + TPB - 1) / TPB, TPB, 0, stream>>>(
      agg, NNODE * DLAT);

  edge_mlp_kernel<<<NEDGE / 64, 128, 0, stream>>>(
      x, edge_attr, eidx, ebW1t, eb_b1, ebW2t, eb_b2, ebW3t, eb_b3,
      eb_g, eb_be, out_edge, agg);

  node_mlp_kernel<<<(NNODE + 63) / 64, 128, 0, stream>>>(
      x, agg, nbW1t, nb_b1, nbW2t, nb_b2, nbW3t, nb_b3,
      nb_g, nb_be, x_out);
}
